// DAttentionBaseline_74775380623540
// MI455X (gfx1250) — matrix operand order repaired and checked by reading, not compiled, not run on hardware
//
#include <hip/hip_runtime.h>
#include <hip/hip_bf16.h>

typedef __attribute__((ext_vector_type(16))) _Float16 v16h;
typedef __attribute__((ext_vector_type(8)))  _Float16 v8h;
typedef __attribute__((ext_vector_type(8)))  float    v8f;

#define CIN 384
#define HW  3136
#define NKV 784
#define NPAD 832

// A-fragment k-permutation within a 32-wide K block:
// half 0 reads k {0..7,16..23} as contiguous bytes, half 1 reads {8..15,24..31}.
__device__ __forceinline__ int perm32(int k) {
    int g = k >> 3;
    return (g == 1) ? k + 8 : (g == 2) ? k - 8 : k;
}

__device__ __forceinline__ unsigned pk2(float a, float b) {
    union { _Float16 h[2]; unsigned u; } z;
    z.h[0] = (_Float16)a; z.h[1] = (_Float16)b;
    return z.u;
}

// 16 contiguous halves (32B, 16B-aligned) -> two ds_load_b128
__device__ __forceinline__ v16h ldfrag(const _Float16* p) {
    v8h lo = *(const v8h*)p;
    v8h hi = *(const v8h*)(p + 8);
    return __builtin_shufflevector(lo, hi, 0,1,2,3,4,5,6,7,8,9,10,11,12,13,14,15);
}

// ---------------------------------------------------------------------------
// WMMA GEMM: Y[b] = W(384x384) * X[b](384 x ldX) + bias
// 64x64 tile, 4 waves x (32x32), K staged in 64-chunks as f16 fragment-order LDS.
// transT: store output transposed per head: [b][h][n][48] (for attention K).
// ---------------------------------------------------------------------------
__global__ __launch_bounds__(128)
void gemm384_wmma(const float* __restrict__ W, const float* __restrict__ bias,
                  const float* __restrict__ X, float* __restrict__ Y,
                  int N, int ldX, int ldY, int transT)
{
    __shared__ _Float16 As16[64 * 64];   // [m][k] rows 128B
    __shared__ _Float16 Bs16[64 * 72];   // [n][k] rows 144B (padded)

    const int b  = blockIdx.z;
    const int m0 = blockIdx.y * 64;
    const int n0 = blockIdx.x * 64;
    const int t    = threadIdx.x;
    const int wave = t >> 5;
    const int lane = t & 31;
    const int half = lane >> 4;
    const int l15  = lane & 15;
    const int mw = (wave >> 1) * 32;
    const int nw = (wave & 1) * 32;

    const float* Xb = X + (size_t)b * CIN * ldX;

    v8f acc[2][2] = {};

    for (int kk = 0; kk < CIN; kk += 64) {
        // stage A (64m x 64k) as f16 pairs in fragment order
        for (int i = t; i < 2048; i += 128) {
            int m  = i >> 5;
            int kp = (i & 31) * 2;
            float2 wv = *(const float2*)&W[(size_t)(m0 + m) * CIN + kk + kp];
            int p = kp;
            *(unsigned*)&As16[m * 64 + p] = pk2(wv.x, wv.y);
        }
        // stage B transposed (64n x 64k), zero outside logical N
        for (int i = t; i < 4096; i += 128) {
            int k = i >> 6, c = i & 63;
            int n = n0 + c;
            float xv = (n < N) ? Xb[(size_t)(kk + k) * ldX + n] : 0.0f;
            Bs16[c * 72 + k] = (_Float16)xv;
        }
        if (kk + 64 < CIN)
            __builtin_prefetch(&Xb[(size_t)(kk + 64) * ldX + n0 + (t & 63)], 0, 1);
        __syncthreads();

        #pragma unroll
        for (int ks = 0; ks < 2; ++ks) {
            v16h af[2], bf[2];
            #pragma unroll
            for (int mi = 0; mi < 2; ++mi)
                af[mi] = ldfrag(&As16[(mw + mi * 16 + l15) * 64 + ks * 32 + half * 16]);
            #pragma unroll
            for (int ni = 0; ni < 2; ++ni)
                bf[ni] = ldfrag(&Bs16[(nw + ni * 16 + l15) * 72 + ks * 32 + half * 16]);
            #pragma unroll
            for (int mi = 0; mi < 2; ++mi)
                #pragma unroll
                for (int ni = 0; ni < 2; ++ni)
                    acc[mi][ni] = __builtin_amdgcn_wmma_f32_16x16x32_f16(
                        false, af[mi], false, bf[ni], (short)0, acc[mi][ni], false, false);
        }
        __syncthreads();
    }

    // store: C/D element r -> M = r + half*8, N = l15
    #pragma unroll
    for (int mi = 0; mi < 2; ++mi)
        #pragma unroll
        for (int ni = 0; ni < 2; ++ni) {
            int n = n0 + nw + ni * 16 + l15;
            if (n < ldY) {
                #pragma unroll
                for (int r = 0; r < 8; ++r) {
                    int m = m0 + mw + mi * 16 + r + half * 8;
                    float v = (n < N) ? (acc[mi][ni][r] + bias[m]) : 0.0f;
                    if (!transT) {
                        Y[(size_t)b * CIN * ldY + (size_t)m * ldY + n] = v;
                    } else {
                        int hh = m / 48, cc = m - hh * 48;   // head, channel-in-head
                        Y[(size_t)b * CIN * ldY + (size_t)hh * 48 * ldY
                          + (size_t)n * 48 + cc] = v;
                    }
                }
            }
        }
}

// ---------------------------------------------------------------------------
// Offset net: depthwise 3x3 s2 -> channel LN -> GELU -> 1x1 to 2ch -> tanh/27 + ref
// ---------------------------------------------------------------------------
__global__ __launch_bounds__(128)
void offset_net_kernel(const float* __restrict__ q, const float* __restrict__ dww,
                       const float* __restrict__ dwb, const float* __restrict__ lng,
                       const float* __restrict__ lnb, const float* __restrict__ pww,
                       float* __restrict__ pos_ws, float* __restrict__ out_pos,
                       float* __restrict__ out_ref)
{
    __shared__ float red[128];
    __shared__ float red2[128];

    const int i  = blockIdx.x;        // 0..783
    const int bg = blockIdx.y;        // 0..31
    const int oy = i / 28, ox = i % 28;
    const int t  = threadIdx.x;

    float hc = 0.f;
    if (t < 96) {
        int b = bg >> 2, g = bg & 3;
        const float* qc = q + (size_t)(b * CIN + g * 96 + t) * HW;
        float a = dwb[t];
        #pragma unroll
        for (int ky = 0; ky < 3; ++ky) {
            int yin = oy * 2 + ky - 1;
            if (yin < 0 || yin > 55) continue;
            #pragma unroll
            for (int kx = 0; kx < 3; ++kx) {
                int xin = ox * 2 + kx - 1;
                if (xin < 0 || xin > 55) continue;
                a += dww[t * 9 + ky * 3 + kx] * qc[yin * 56 + xin];
            }
        }
        hc = a;
    }
    red[t]  = (t < 96) ? hc : 0.f;
    red2[t] = (t < 96) ? hc * hc : 0.f;
    __syncthreads();
    for (int s = 64; s > 0; s >>= 1) {
        if (t < s) { red[t] += red[t + s]; red2[t] += red2[t + s]; }
        __syncthreads();
    }
    float mu  = red[0] * (1.f / 96.f);
    float var = red2[0] * (1.f / 96.f) - mu * mu;
    float inv = rsqrtf(var + 1e-5f);
    __syncthreads();

    float p0 = 0.f, p1 = 0.f;
    if (t < 96) {
        float hn = (hc - mu) * inv * lng[t] + lnb[t];
        float ge = 0.5f * hn * (1.f + erff(hn * 0.7071067811865476f));
        p0 = pww[t] * ge;
        p1 = pww[96 + t] * ge;
    }
    red[t] = p0; red2[t] = p1;
    __syncthreads();
    for (int s = 64; s > 0; s >>= 1) {
        if (t < s) { red[t] += red[t + s]; red2[t] += red2[t + s]; }
        __syncthreads();
    }
    if (t == 0) {
        float offy = tanhf(red[0])  * (1.f / 27.f);
        float offx = tanhf(red2[0]) * (1.f / 27.f);
        float ry = (0.5f + (float)oy) * (2.f / 27.f) - 1.f;
        float rx = (0.5f + (float)ox) * (2.f / 27.f) - 1.f;
        float py = offy + ry, px = offx + rx;
        size_t o = ((size_t)bg * NKV + i) * 2;
        pos_ws[o] = py;  pos_ws[o + 1] = px;
        out_pos[o] = py; out_pos[o + 1] = px;
        out_ref[o] = ry; out_ref[o + 1] = rx;
    }
}

// ---------------------------------------------------------------------------
// Bilinear grid sample (align_corners=True, zero padding)
// ---------------------------------------------------------------------------
__global__ __launch_bounds__(256)
void grid_sample_kernel(const float* __restrict__ x, const float* __restrict__ pos_ws,
                        float* __restrict__ xs)
{
    int tid = blockIdx.x * 256 + threadIdx.x;
    if (tid >= 32 * 96 * NKV) return;
    int i  = tid % NKV;
    int c  = (tid / NKV) % 96;
    int bg = tid / (NKV * 96);

    float py = pos_ws[((size_t)bg * NKV + i) * 2];
    float px = pos_ws[((size_t)bg * NKV + i) * 2 + 1];
    float gx = (px + 1.f) * 0.5f * 55.f;
    float gy = (py + 1.f) * 0.5f * 55.f;
    float x0 = floorf(gx), y0 = floorf(gy);
    float wx1 = gx - x0, wy1 = gy - y0;
    float wx0 = 1.f - wx1, wy0 = 1.f - wy1;

    int b = bg >> 2, g = bg & 3;
    const float* img = x + (size_t)(b * CIN + g * 96 + c) * HW;
    auto samp = [&](float fx, float fy) -> float {
        if (fx < 0.f || fx > 55.f || fy < 0.f || fy > 55.f) return 0.f;
        return img[(int)fy * 56 + (int)fx];
    };
    float v00 = samp(x0,       y0);
    float v01 = samp(x0 + 1.f, y0);
    float v10 = samp(x0,       y0 + 1.f);
    float v11 = samp(x0 + 1.f, y0 + 1.f);
    float r = v00 * wx0 * wy0 + v01 * wx1 * wy0 + v10 * wx0 * wy1 + v11 * wx1 * wy1;
    xs[(size_t)(b * CIN + g * 96 + c) * NKV + i] = r;
}

// ---------------------------------------------------------------------------
// Flash attention per (b,h). kT is pre-transposed [bh][n(832)][c(48)].
// All operands staged in LDS as f16 in fragment order -> 2x ds_load_b128/frag.
// ---------------------------------------------------------------------------
__global__ __launch_bounds__(128)
void flash_attn_kernel(const float* __restrict__ q, const float* __restrict__ kT,
                       const float* __restrict__ vbuf, float* __restrict__ outb)
{
    __shared__ _Float16 Qs16[64 * 64];       // [m][c padded 64]
    __shared__ _Float16 Ks16[64 * 64];       // [n][c padded 64]
    __shared__ _Float16 Vs16[48 * 64];       // [c][n]
    __shared__ _Float16 Pf[4][16 * 64];      // per-wave P rows, A-frag order
    __shared__ float    SPf[4][16][66];      // raw S scores (f32)
    __shared__ float    alphas[4][16];
    __shared__ float    lfin[4][16];

    const int bh = blockIdx.y;
    const int b  = bh >> 3, h = bh & 7;
    const int m0 = blockIdx.x * 64;
    const int t    = threadIdx.x;
    const int wave = t >> 5;
    const int lane = t & 31;
    const int half = lane >> 4;
    const int l15  = lane & 15;

    const float* qb = q    + (size_t)(b * CIN + h * 48) * HW;
    const float* kb = kT   + (size_t)bh * NPAD * 48;
    const float* vb = vbuf + (size_t)(b * CIN + h * 48) * NPAD;
    float*       ob = outb + (size_t)(b * CIN + h * 48) * HW;

    const float scale = 0.14433756729740643f;  // 1/sqrt(48)

    // stage Q (A-frag order, c padded 48->64 with zeros)
    for (int i = t; i < 2048; i += 128) {
        int cp = i >> 6, m = i & 63;
        int c = cp * 2;
        float a  = (c     < 48) ? qb[(size_t)c       * HW + m0 + m] * scale : 0.f;
        float b2 = (c + 1 < 48) ? qb[(size_t)(c + 1) * HW + m0 + m] * scale : 0.f;
        int p = c;
        *(unsigned*)&Qs16[m * 64 + p] = pk2(a, b2);
    }
    __syncthreads();

    v16h aq[2];
    #pragma unroll
    for (int s = 0; s < 2; ++s)
        aq[s] = ldfrag(&Qs16[(wave * 16 + l15) * 64 + s * 32 + half * 16]);

    float m_i = -3.0e38f, l_i = 0.f;
    v8f oacc[3] = {};

    for (int chunk = 0; chunk < 13; ++chunk) {
        int n0 = chunk * 64;
        // stage K (row copies from transposed buffer) + zero pad c 48..63
        for (int i = t; i < 64 * 24; i += 128) {
            int n = i / 24, cp = i % 24;
            float2 kv = *(const float2*)&kb[(size_t)(n0 + n) * 48 + cp * 2];
            *(unsigned*)&Ks16[n * 64 + cp * 2] = pk2(kv.x, kv.y);
        }
        for (int i = t; i < 512; i += 128) {
            int n = i >> 3, j = i & 7;
            *(unsigned*)&Ks16[n * 64 + 48 + j * 2] = 0u;
        }
        // stage V (natural row copies)
        for (int i = t; i < 1536; i += 128) {
            int c = i >> 5, np = (i & 31) * 2;
            float2 vv = *(const float2*)&vb[(size_t)c * NPAD + n0 + np];
            *(unsigned*)&Vs16[c * 64 + np] = pk2(vv.x, vv.y);
        }
        __syncthreads();

        // S tile (16 x 64 per wave): 8 WMMA
        #pragma unroll
        for (int nf = 0; nf < 4; ++nf) {
            v8f s = {};
            #pragma unroll
            for (int ks = 0; ks < 2; ++ks) {
                v16h bkf = ldfrag(&Ks16[(nf * 16 + l15) * 64 + ks * 32 + half * 16]);
                s = __builtin_amdgcn_wmma_f32_16x16x32_f16(
                        false, aq[ks], false, bkf, (short)0, s, false, false);
            }
            #pragma unroll
            for (int r = 0; r < 8; ++r)
                SPf[wave][r + half * 8][nf * 16 + l15] = s[r];
        }
        __syncthreads();

        // online softmax, one lane per query row; writes P as f16 A-frag order
        if (lane < 16) {
            int row = lane;
            int nvalid = 784 - n0; if (nvalid > 64) nvalid = 64;
            float mc = -3.0e38f;
            for (int j = 0; j < nvalid; ++j) mc = fmaxf(mc, SPf[wave][row][j]);
            float mnew  = fmaxf(m_i, mc);
            float alpha = __expf(m_i - mnew);
            float ps = 0.f;
            for (int j2 = 0; j2 < 32; ++j2) {
                int j = j2 * 2;
                float p0 = (j     < nvalid) ? __expf(SPf[wave][row][j]     - mnew) : 0.f;
                float p1 = (j + 1 < nvalid) ? __expf(SPf[wave][row][j + 1] - mnew) : 0.f;
                int pa = j;
                *(unsigned*)&Pf[wave][row * 64 + pa] = pk2(p0, p1);
                ps += p0 + p1;
            }
            l_i = l_i * alpha + ps;
            m_i = mnew;
            alphas[wave][row] = alpha;
        }
        __syncthreads();

        // rescale O accumulators
        #pragma unroll
        for (int j = 0; j < 3; ++j)
            #pragma unroll
            for (int r = 0; r < 8; ++r)
                oacc[j][r] *= alphas[wave][r + half * 8];

        // O += P * V^T : 6 WMMA
        v16h ap[2];
        #pragma unroll
        for (int s = 0; s < 2; ++s)
            ap[s] = ldfrag(&Pf[wave][l15 * 64 + s * 32 + half * 16]);
        #pragma unroll
        for (int cf = 0; cf < 3; ++cf)
            #pragma unroll
            for (int ks = 0; ks < 2; ++ks) {
                v16h bvf = ldfrag(&Vs16[(cf * 16 + l15) * 64 + ks * 32 + half * 16]);
                oacc[cf] = __builtin_amdgcn_wmma_f32_16x16x32_f16(
                               false, ap[ks], false, bvf, (short)0, oacc[cf], false, false);
            }
        __syncthreads();
    }

    if (lane < 16) lfin[wave][lane] = l_i;
    __syncthreads();

    #pragma unroll
    for (int cf = 0; cf < 3; ++cf) {
        int c = cf * 16 + l15;
        #pragma unroll
        for (int r = 0; r < 8; ++r) {
            int ml = r + half * 8;
            float v = oacc[cf][r] / lfin[wave][ml];
            ob[(size_t)c * HW + m0 + wave * 16 + ml] = v;
        }
    }
}

// ---------------------------------------------------------------------------
extern "C" void kernel_launch(void* const* d_in, const int* in_sizes, int n_in,
                              void* d_out, int out_size, void* d_ws, size_t ws_size,
                              hipStream_t stream)
{
    (void)in_sizes; (void)n_in; (void)out_size; (void)ws_size;

    const float* x   = (const float*)d_in[0];
    const float* Wq  = (const float*)d_in[1];
    const float* bq  = (const float*)d_in[2];
    const float* Wk  = (const float*)d_in[3];
    const float* bk  = (const float*)d_in[4];
    const float* Wv  = (const float*)d_in[5];
    const float* bv  = (const float*)d_in[6];
    const float* Wo  = (const float*)d_in[7];
    const float* bo  = (const float*)d_in[8];
    const float* dww = (const float*)d_in[9];
    const float* dwb = (const float*)d_in[10];
    const float* lng = (const float*)d_in[11];
    const float* lnb = (const float*)d_in[12];
    const float* pww = (const float*)d_in[13];

    float* ws      = (float*)d_ws;
    float* q       = ws;                    // 8*384*3136
    float* xs      = q + 9633792;           // 8*384*784
    float* kbuf    = xs + 2408448;          // 8*8 heads * 832 * 48 (transposed)
    float* vbuf    = kbuf + 2555904;        // 8*384*832
    float* attnout = vbuf + 2555904;        // 8*384*3136
    float* posws   = attnout + 9633792;     // 32*784*2

    float* y_out   = (float*)d_out;         // 8*384*3136
    float* pos_out = y_out + 9633792;       // 8*4*28*28*2
    float* ref_out = pos_out + 50176;       // 8*4*28*28*2

    dim3 blk(128);
    // q = Wq x + bq
    gemm384_wmma<<<dim3(49, 6, 8), blk, 0, stream>>>(Wq, bq, x, q, HW, HW, HW, 0);
    // offset net -> pos (+ d_out pos/ref slices)
    offset_net_kernel<<<dim3(784, 32), blk, 0, stream>>>(q, dww, dwb, lng, lnb, pww,
                                                         posws, pos_out, ref_out);
    // deformed sampling
    grid_sample_kernel<<<dim3((32 * 96 * NKV + 255) / 256), dim3(256), 0, stream>>>(x, posws, xs);
    // k transposed per head, v natural; both padded to 832 cols with zeros
    gemm384_wmma<<<dim3(13, 6, 8), blk, 0, stream>>>(Wk, bk, xs, kbuf, NKV, NKV, NPAD, 1);
    gemm384_wmma<<<dim3(13, 6, 8), blk, 0, stream>>>(Wv, bv, xs, vbuf, NKV, NKV, NPAD, 0);
    // flash attention
    flash_attn_kernel<<<dim3(49, 64), blk, 0, stream>>>(q, kbuf, vbuf, attnout);
    // y = Wo out + bo
    gemm384_wmma<<<dim3(49, 6, 8), blk, 0, stream>>>(Wo, bo, attnout, y_out, HW, HW, HW, 0);
}
